// MoELoRADelta_26431228740180
// MI455X (gfx1250) — compile-verified, matrix operand order checked by reading, not run on hardware
//
#include <hip/hip_runtime.h>
#include <hip/hip_bf16.h>
#include <stdint.h>
#include <math.h>

// MoE LoRA delta, MI455X / gfx1250, wave32, v_wmma_f32_16x16x32_f16 + TDM.
// Shapes: T=8192, D=4096, E=4, R=16, O=4096.
//   K1: split-K x-pass; weights staged to LDS by the Tensor Data Mover
//       (double-buffered, TENSORcnt-synced), x streamed via global_load_b128.
//   K2: deterministic split-K reduction.
//   K3: per-lane top-2 softmax gates folded into A-fragment; delta = w @ Bcat^T.

typedef __attribute__((ext_vector_type(16))) _Float16 v16h;
typedef __attribute__((ext_vector_type(8)))  float    v8f;
typedef __attribute__((ext_vector_type(4)))  unsigned int v4u;
typedef __attribute__((ext_vector_type(8)))  int      v8i;
typedef __attribute__((ext_vector_type(4)))  int      v4i;

#define LANE   (threadIdx.x & 31)
#define WAVE   (threadIdx.x >> 5)

__device__ __forceinline__ v8f wmma_f16(v16h a, v16h b, v8f c) {
    // (neg_a, A, neg_b, B, c_mod, C, reuse_a, reuse_b)
    return __builtin_amdgcn_wmma_f32_16x16x32_f16(false, a, false, b, (short)0, c, false, false);
}

// Load 8 contiguous f32 (32B aligned, global or LDS) into f16 slots [OFF..OFF+7].
template <int OFF>
__device__ __forceinline__ void cvt8(const float* __restrict__ p, v16h& d) {
    const float4* q = reinterpret_cast<const float4*>(p);
    float4 a = q[0], b = q[1];
    d[OFF + 0] = (_Float16)a.x; d[OFF + 1] = (_Float16)a.y;
    d[OFF + 2] = (_Float16)a.z; d[OFF + 3] = (_Float16)a.w;
    d[OFF + 4] = (_Float16)b.x; d[OFF + 5] = (_Float16)b.y;
    d[OFF + 6] = (_Float16)b.z; d[OFF + 7] = (_Float16)b.w;
}

template <int OFF>
__device__ __forceinline__ void cvt8s(const float* __restrict__ p, float s, v16h& d) {
    const float4* q = reinterpret_cast<const float4*>(p);
    float4 a = q[0], b = q[1];
    d[OFF + 0] = (_Float16)(a.x * s); d[OFF + 1] = (_Float16)(a.y * s);
    d[OFF + 2] = (_Float16)(a.z * s); d[OFF + 3] = (_Float16)(a.w * s);
    d[OFF + 4] = (_Float16)(b.x * s); d[OFF + 5] = (_Float16)(b.y * s);
    d[OFF + 6] = (_Float16)(b.z * s); d[OFF + 7] = (_Float16)(b.w * s);
}

// ---------------------------------------------------------------------------
// TDM: async DMA of a 2-D f32 tile (rows x 32 elems, row stride = stride_elems)
// from global memory into LDS at byte offset lds_off. D# per ISA ch.8.
// ---------------------------------------------------------------------------
__device__ __forceinline__ void tdm_load_2d(const float* gsrc, unsigned lds_off,
                                            unsigned rows, unsigned tile_w,
                                            unsigned stride_elems) {
    unsigned long long ga = (unsigned long long)(uintptr_t)gsrc;
    v4u g0;
    g0[0] = 1u;                                            // count=1, user mode
    g0[1] = lds_off;                                       // lds_addr (bytes)
    g0[2] = (unsigned)(ga & 0xFFFFFFFFull);                // global_addr[31:0]
    g0[3] = (unsigned)((ga >> 32) & 0x01FFFFFFull)         // global_addr[56:32]
            | (2u << 30);                                  // type=2 ("image")
    v8i g1;
    g1[0] = (int)(2u << 16);                               // data_size=2 -> 4 bytes
    g1[1] = (int)((stride_elems & 0xFFFFu) << 16);         // tensor_dim0[15:0]
    g1[2] = (int)(((stride_elems >> 16) & 0xFFFFu)         // tensor_dim0[31:16]
            | ((rows & 0xFFFFu) << 16));                   // tensor_dim1[15:0]
    g1[3] = (int)(((rows >> 16) & 0xFFFFu)                 // tensor_dim1[31:16]
            | ((tile_w & 0xFFFFu) << 16));                 // tile_dim0
    g1[4] = (int)(rows & 0xFFFFu);                         // tile_dim1 (tile_dim2=0)
    g1[5] = (int)stride_elems;                             // tensor_dim0_stride[31:0]
    g1[6] = 0;                                             // stride hi / dim1_stride lo
    g1[7] = 0;
    v4i z4 = {0, 0, 0, 0};
#if __clang_major__ >= 23
    v8i z8 = {0, 0, 0, 0, 0, 0, 0, 0};
    __builtin_amdgcn_tensor_load_to_lds(g0, g1, z4, z4, z8, 0);
#else
    __builtin_amdgcn_tensor_load_to_lds(g0, g1, z4, z4, 0);
#endif
}

// ---------------------------------------------------------------------------
// Kernel 1: fused router + mid. One wave = one 16-token tile, split-K over D.
// Weight K-chunks (Acat 64x32 + router_w 4x32, f32) staged to LDS via TDM,
// double-buffered; x streamed from global. 5 WMMA tiles per 32-K step.
// ---------------------------------------------------------------------------
__global__ __launch_bounds__(256) void moe_mid_router(
    const float* __restrict__ x,     // (T, D)
    const float* __restrict__ rw,    // (E=4, D)
    const float* __restrict__ A,     // (E, R, D) == Acat(64, D), row = e*16+r
    float* __restrict__ midp,        // (splitk, T, 64)
    float* __restrict__ logp,        // (splitk, T, 4)
    int T) {
    constexpr int D = 4096;
    __shared__ __align__(16) float smem[2][68 * 32];   // [buf][row*32 + col]

    const int lane = LANE, half = lane >> 4, m = lane & 15;
    const int t0   = (blockIdx.x * 8 + WAVE) * 16;
    const int kseg = D / gridDim.y;
    const int ks   = blockIdx.y * kseg;
    const int nchunk = kseg / 32;

    const unsigned lds_base = (unsigned)(uintptr_t)&smem[0][0];
    const unsigned buf_bytes = 68 * 32 * 4;

    // stage chunk 0 into buffer 0 (wave 0 issues; TDM ignores EXEC)
    if (WAVE == 0) {
        tdm_load_2d(A  + ks, lds_base,                64, 32, D);
        tdm_load_2d(rw + ks, lds_base + 64 * 32 * 4,   4, 32, D);
        __builtin_amdgcn_s_wait_tensorcnt(0);
    }
    __syncthreads();

    int rowoff[5];
#pragma unroll
    for (int nt = 0; nt < 4; ++nt) rowoff[nt] = (nt * 16 + m) * 32;
    rowoff[4] = (64 + (m & 3)) * 32;    // router rows; cols>=4 never stored

    v8f acc[5];
#pragma unroll
    for (int nt = 0; nt < 5; ++nt) acc[nt] = (v8f){0.f,0.f,0.f,0.f,0.f,0.f,0.f,0.f};

    const float* xrow = x + (size_t)(t0 + m) * D;

    for (int kc = 0; kc < nchunk; ++kc) {
        const int k = ks + kc * 32;
        const float* sb = &smem[kc & 1][0];

        // kick off DMA for the next chunk into the other buffer
        if (WAVE == 0 && kc + 1 < nchunk) {
            const unsigned nb = lds_base + ((kc + 1) & 1) * buf_bytes;
            tdm_load_2d(A  + ks + (kc + 1) * 32, nb,                64, 32, D);
            tdm_load_2d(rw + ks + (kc + 1) * 32, nb + 64 * 32 * 4,   4, 32, D);
        }

        __builtin_prefetch(xrow + k + 128, 0, 3);      // global_prefetch_b8 on x
        v16h af;                                        // A frag: 16x32 f16 layout
        cvt8<0>(xrow + k + half * 8,      af);
        cvt8<8>(xrow + k + 16 + half * 8, af);
#pragma unroll
        for (int nt = 0; nt < 5; ++nt) {
            v16h bf;                                    // B[k][n] from LDS
            cvt8<0>(sb + rowoff[nt] + half * 8,      bf);
            cvt8<8>(sb + rowoff[nt] + 16 + half * 8, bf);
            acc[nt] = wmma_f16(af, bf, acc[nt]);
        }

        if (WAVE == 0 && kc + 1 < nchunk) __builtin_amdgcn_s_wait_tensorcnt(0);
        __syncthreads();   // next-chunk data ready; all readers done with old buf
    }

    // C/D layout: acc[v] = C[row = v + 8*half][col = m]
    const size_t mbase = ((size_t)blockIdx.y * T + t0) * 64;
#pragma unroll
    for (int nt = 0; nt < 4; ++nt)
#pragma unroll
        for (int v = 0; v < 8; ++v)
            midp[mbase + (size_t)(v + half * 8) * 64 + nt * 16 + m] = acc[nt][v];

    if (m < 4) {
        const size_t lbase = ((size_t)blockIdx.y * T + t0) * 4;
#pragma unroll
        for (int v = 0; v < 8; ++v)
            logp[lbase + (size_t)(v + half * 8) * 4 + m] = acc[4][v];
    }
}

// ---------------------------------------------------------------------------
// Kernel 2: deterministic split-K reduction (fixed summation order).
// ---------------------------------------------------------------------------
__global__ __launch_bounds__(256) void moe_reduce(
    const float* __restrict__ midp, const float* __restrict__ logp,
    float* __restrict__ mid, float* __restrict__ logits, int T, int splitk) {
    const int i  = blockIdx.x * blockDim.x + threadIdx.x;
    const int n1 = T * 64;
    if (i < n1) {
        float s = 0.f;
        for (int y = 0; y < splitk; ++y) s += midp[(size_t)y * n1 + i];
        mid[i] = s;
    } else if (i < n1 + T * 4) {
        const int j = i - n1;
        float s = 0.f;
        for (int y = 0; y < splitk; ++y) s += logp[(size_t)y * T * 4 + j];
        logits[j] = s;
    }
}

// ---------------------------------------------------------------------------
// Kernel 3: gates (top-2 softmax of E=4, per lane) + delta = w @ Bcat^T, K=64.
// Bcat[o][e*16+r] = B[e][o][r]; gate*LORA_SCALE folded into A fragment.
// ---------------------------------------------------------------------------
__global__ __launch_bounds__(256) void moe_out(
    const float* __restrict__ mid,    // (T, 64)
    const float* __restrict__ logits, // (T, 4)
    const float* __restrict__ Bw,     // (E=4, O, R=16)
    float* __restrict__ out,          // (T, O)
    int T) {
    constexpr int O = 4096, R = 16;
    const int lane = LANE, half = lane >> 4, m = lane & 15;
    const int t0 = (blockIdx.x * 8 + WAVE) * 16;
    const int t  = t0 + m;

    // top-2 softmax over 4 logits (duplicated across the two lane halves)
    float4 lg = *reinterpret_cast<const float4*>(logits + (size_t)t * 4);
    float l[4] = {lg.x, lg.y, lg.z, lg.w};
    int i0 = 0;
#pragma unroll
    for (int i = 1; i < 4; ++i) if (l[i] > l[i0]) i0 = i;
    int i1 = (i0 == 0) ? 1 : 0;
#pragma unroll
    for (int i = 0; i < 4; ++i) if (i != i0 && l[i] > l[i1]) i1 = i;
    const float eb  = __expf(l[i1] - l[i0]);
    const float inv = 1.0f / (1.0f + eb);
    float g[4] = {0.f, 0.f, 0.f, 0.f};
    g[i0] = inv * 4.0f;          // LORA_SCALE = 16/sqrt(16) = 4
    g[i1] = eb * inv * 4.0f;

    // A fragments: w[t][k] = mid[t][k] * g[k>>4]; each 8-group is one expert
    const float* mrow = mid + (size_t)t * 64;
    v16h af[2];
#pragma unroll
    for (int kc = 0; kc < 2; ++kc) {
        cvt8s<0>(mrow + kc * 32 + half * 8,      g[kc * 2],     af[kc]); // e = 2*kc
        cvt8s<8>(mrow + kc * 32 + 16 + half * 8, g[kc * 2 + 1], af[kc]); // e = 2*kc+1
    }

    const int per = (O / 16) / gridDim.y;   // N tiles per block (y-split)
    const int nt0 = blockIdx.y * per;
    for (int n = 0; n < per; ++n) {
        const int nt = nt0 + n;
        const int o  = nt * 16 + m;         // this lane's output column
        v8f acc = (v8f){0.f,0.f,0.f,0.f,0.f,0.f,0.f,0.f};
#pragma unroll
        for (int kc = 0; kc < 2; ++kc) {
            v16h bf;   // B[k][n] = Bw[e = k>>4][o][r = k&15], r = half*8 + 0..7
            cvt8<0>(Bw + ((size_t)(kc * 2)     * O + o) * R + half * 8, bf);
            cvt8<8>(Bw + ((size_t)(kc * 2 + 1) * O + o) * R + half * 8, bf);
            acc = wmma_f16(af[kc], bf, acc);
        }
        float* op = out + (size_t)t0 * O + nt * 16 + m;
#pragma unroll
        for (int v = 0; v < 8; ++v)
            op[(size_t)(v + half * 8) * O] = acc[v];
    }
}

// ---------------------------------------------------------------------------
extern "C" void kernel_launch(void* const* d_in, const int* in_sizes, int n_in,
                              void* d_out, int out_size, void* d_ws, size_t ws_size,
                              hipStream_t stream) {
    constexpr int D = 4096;
    const float* x   = (const float*)d_in[0];
    const float* rw  = (const float*)d_in[1];
    const float* A   = (const float*)d_in[2];
    const float* Bw  = (const float*)d_in[3];
    float*       out = (float*)d_out;
    const int T = in_sizes[0] / D;          // 8192

    // split-K factor sized against workspace: (splitk+1) * T * 68 floats needed
    int splitk = 8;
    while (splitk > 1 && (size_t)(splitk + 1) * (size_t)T * 68 * 4 > ws_size) splitk >>= 1;

    char* ws = (char*)d_ws;
    float* midp   = (float*)ws;                                             // splitk*T*64
    float* logp   = (float*)(ws + (size_t)splitk * T * 64 * 4);             // splitk*T*4
    float* mid    = (float*)(ws + (size_t)splitk * T * 68 * 4);             // T*64
    float* logits = (float*)(ws + (size_t)splitk * T * 68 * 4 + (size_t)T * 64 * 4); // T*4

    dim3 blk(256);
    dim3 g1(T / 128, splitk);
    moe_mid_router<<<g1, blk, 0, stream>>>(x, rw, A, midp, logp, T);

    const int nred = T * 68;
    moe_reduce<<<(nred + 255) / 256, blk, 0, stream>>>(midp, logp, mid, logits, T, splitk);

    dim3 g3(T / 128, 8);                    // 8-way N split: 512 blocks total
    moe_out<<<g3, blk, 0, stream>>>(mid, logits, Bw, out, T);
}
